// GNN_layer_GAT_8589934592117
// MI455X (gfx1250) — matrix operand order repaired and checked by reading, not compiled, not run on hardware
//
#include <hip/hip_runtime.h>

// ---------------------------------------------------------------------------
// Problem constants (match reference)
// ---------------------------------------------------------------------------
#define NN     2048            // nodes
#define FF     3072            // feature length L == F
#define SLOPE  0.2f
#define BNEPS  1e-5f

typedef __bf16 v16bf __attribute__((ext_vector_type(16)));
typedef float  v8f   __attribute__((ext_vector_type(8)));

union Frag16 { v16bf v; uint4 q[2]; };

// ---------------------------------------------------------------------------
// K0: convert + transpose gat weights:  wt[c][n][k] = (bf16) W_c[k][n]
// block (32,8), grid (FF/32, FF/32, 2)
// ---------------------------------------------------------------------------
__global__ __launch_bounds__(256)
void wcvt_kernel(const float* __restrict__ w1, const float* __restrict__ w2,
                 unsigned short* __restrict__ wt_raw)
{
    __shared__ float tile[32][33];
    const int c  = blockIdx.z;
    const float* W = (c == 0) ? w1 : w2;
    __bf16* wt = (__bf16*)wt_raw + (size_t)c * FF * FF;

    const int nb = blockIdx.x * 32;
    const int kb = blockIdx.y * 32;
    for (int ty = threadIdx.y; ty < 32; ty += 8)
        tile[ty][threadIdx.x] = W[(size_t)(kb + ty) * FF + nb + threadIdx.x];
    __syncthreads();
    for (int ty = threadIdx.y; ty < 32; ty += 8)
        wt[(size_t)(nb + ty) * FF + kb + threadIdx.x] = (__bf16)tile[threadIdx.x][ty];
}

// ---------------------------------------------------------------------------
// K1: conv1(k7)+BN+ReLU -> conv2(k7)+BN+ReLU -> roll
//     writes fp32 temp_value [n][c][l]  and bf16 tv [c][n][l] (GEMM A)
// one block per node, 256 threads
// ---------------------------------------------------------------------------
__global__ __launch_bounds__(256)
void conv_roll_kernel(const float* __restrict__ x,
                      const float* __restrict__ dtp, const float* __restrict__ dts,
                      const float* __restrict__ w1, const float* __restrict__ b1,
                      const float* __restrict__ g1, const float* __restrict__ be1,
                      const float* __restrict__ m1, const float* __restrict__ v1,
                      const float* __restrict__ w2, const float* __restrict__ b2,
                      const float* __restrict__ g2, const float* __restrict__ be2,
                      const float* __restrict__ m2, const float* __restrict__ v2,
                      float* __restrict__ temp_value,
                      unsigned short* __restrict__ tv_raw)
{
    __shared__ float sx[3 * FF];
    __shared__ float sh[2 * FF];
    const int n   = blockIdx.x;
    const int tid = threadIdx.x;
    __bf16* tvbf = (__bf16*)tv_raw;

    for (int i = tid; i < 3 * FF; i += blockDim.x)
        sx[i] = x[(size_t)n * 3 * FF + i];
    __syncthreads();

    // conv1 + bn1 + relu
    for (int o = 0; o < 2; ++o) {
        const float sc = g1[o] / sqrtf(v1[o] + BNEPS);
        const float mu = m1[o], bt = be1[o], bb = b1[o];
        for (int l = tid; l < FF; l += blockDim.x) {
            float acc = bb;
            #pragma unroll
            for (int i = 0; i < 3; ++i)
                #pragma unroll
                for (int t = 0; t < 7; ++t) {
                    int ll = l + t - 3;
                    if (ll >= 0 && ll < FF)
                        acc += sx[i * FF + ll] * w1[(o * 3 + i) * 7 + t];
                }
            sh[o * FF + l] = fmaxf((acc - mu) * sc + bt, 0.0f);
        }
    }
    __syncthreads();

    const int sp = (int)(dtp[n] * (float)FF);
    const int ss = (int)(dts[n] * (float)FF);

    // conv2 + bn2 + relu + roll
    for (int o = 0; o < 2; ++o) {
        const float sc = g2[o] / sqrtf(v2[o] + BNEPS);
        const float mu = m2[o], bt = be2[o], bb = b2[o];
        const int   s  = (o == 0) ? sp : ss;
        for (int l = tid; l < FF; l += blockDim.x) {
            float acc = bb;
            #pragma unroll
            for (int i = 0; i < 2; ++i)
                #pragma unroll
                for (int t = 0; t < 7; ++t) {
                    int ll = l + t - 3;
                    if (ll >= 0 && ll < FF)
                        acc += sh[i * FF + ll] * w2[(o * 2 + i) * 7 + t];
                }
            float val = fmaxf((acc - mu) * sc + bt, 0.0f);
            int ld = l + s; if (ld >= FF) ld -= FF;          // dest of roll
            temp_value[((size_t)n * 2 + o) * FF + ld] = val;
            tvbf[((size_t)o * NN + n) * FF + ld]      = (__bf16)val;
        }
    }
}

// ---------------------------------------------------------------------------
// K2: WMMA GEMM  h[c] = tv[c] (2048x3072 bf16) @ W_c (3072x3072, via Wt bf16)
// block = 256 thr (8 waves, 2x4 wave grid), block tile 64(M) x 128(N)
// each wave: 32x32 = 2x2 WMMA accumulators -> 2 b128 loads per WMMA
// grid (FF/128, NN/64, 2)
// ---------------------------------------------------------------------------
__global__ __launch_bounds__(256)
void gemm_wmma_kernel(const unsigned short* __restrict__ tv_raw,
                      const unsigned short* __restrict__ wt_raw,
                      float* __restrict__ h)
{
    const int c = blockIdx.z;
    const __bf16* A  = (const __bf16*)tv_raw + (size_t)c * NN * FF;  // [2048][3072]
    const __bf16* Bt = (const __bf16*)wt_raw + (size_t)c * FF * FF;  // [n][k]
    float* H = h + (size_t)c * NN * FF;

    const int tid  = threadIdx.x;
    const int lane = tid & 31;
    const int wv   = tid >> 5;
    const int half = lane >> 4;        // 0: lanes 0-15, 1: lanes 16-31
    const int l16  = lane & 15;
    const int wm   = wv & 1;           // 2 wave-rows
    const int wn   = wv >> 1;          // 4 wave-cols

    const int rowA0 = blockIdx.y * 64 + wm * 32 + l16;   // M tile 0
    const int rowA1 = rowA0 + 16;                        // M tile 1
    const int colB0 = blockIdx.x * 128 + wn * 32 + l16;  // N tile 0
    const int colB1 = colB0 + 16;                        // N tile 1

    const __bf16* aRow0 = A  + (size_t)rowA0 * FF;
    const __bf16* aRow1 = A  + (size_t)rowA1 * FF;
    const __bf16* bRow0 = Bt + (size_t)colB0 * FF;
    const __bf16* bRow1 = Bt + (size_t)colB1 * FF;

    const int aoff = half * 8;         // A frag: K {0..7,16..23} / {8..15,24..31}
    const int boff = half * 8;         // B frag: K {0..7,16..23} / {8..15,24..31}

    v8f c00 = {}; v8f c01 = {}; v8f c10 = {}; v8f c11 = {};
    for (int k0 = 0; k0 < FF; k0 += 32) {
        Frag16 a0, a1, b0, b1;
        a0.q[0] = *(const uint4*)(aRow0 + k0 + aoff);
        a0.q[1] = *(const uint4*)(aRow0 + k0 + aoff + 16);
        a1.q[0] = *(const uint4*)(aRow1 + k0 + aoff);
        a1.q[1] = *(const uint4*)(aRow1 + k0 + aoff + 16);
        b0.q[0] = *(const uint4*)(bRow0 + k0 + boff);
        b0.q[1] = *(const uint4*)(bRow0 + k0 + boff + 16);
        b1.q[0] = *(const uint4*)(bRow1 + k0 + boff);
        b1.q[1] = *(const uint4*)(bRow1 + k0 + boff + 16);
        c00 = __builtin_amdgcn_wmma_f32_16x16x32_bf16(false, a0.v, false, b0.v,
                                                      (short)0, c00, false, false);
        c01 = __builtin_amdgcn_wmma_f32_16x16x32_bf16(false, a0.v, false, b1.v,
                                                      (short)0, c01, false, false);
        c10 = __builtin_amdgcn_wmma_f32_16x16x32_bf16(false, a1.v, false, b0.v,
                                                      (short)0, c10, false, false);
        c11 = __builtin_amdgcn_wmma_f32_16x16x32_bf16(false, a1.v, false, b1.v,
                                                      (short)0, c11, false, false);
    }

    const int mBase = blockIdx.y * 64 + wm * 32 + half * 8;
    #pragma unroll
    for (int r = 0; r < 8; ++r) {
        H[(size_t)(mBase + r)      * FF + colB0] = c00[r];
        H[(size_t)(mBase + r)      * FF + colB1] = c01[r];
        H[(size_t)(mBase + 16 + r) * FF + colB0] = c10[r];
        H[(size_t)(mBase + 16 + r) * FF + colB1] = c11[r];
    }
}

// ---------------------------------------------------------------------------
// K3: attention projections: s_src[c][n] = h[c][n]·a_src_c ; same for a_dst
// grid (NN, 2), block 256
// ---------------------------------------------------------------------------
__global__ __launch_bounds__(256)
void dot_kernel(const float* __restrict__ h,
                const float* __restrict__ as1, const float* __restrict__ ad1,
                const float* __restrict__ as2, const float* __restrict__ ad2,
                float* __restrict__ ssrc, float* __restrict__ sdst)
{
    const int n = blockIdx.x, c = blockIdx.y;
    const float* as = (c == 0) ? as1 : as2;
    const float* ad = (c == 0) ? ad1 : ad2;
    const float* hr = h + ((size_t)c * NN + n) * FF;
    float s0 = 0.f, s1 = 0.f;
    for (int f = threadIdx.x; f < FF; f += blockDim.x) {
        float hv = hr[f];
        s0 += hv * as[f];
        s1 += hv * ad[f];
    }
    __shared__ float r0[256], r1[256];
    r0[threadIdx.x] = s0; r1[threadIdx.x] = s1;
    __syncthreads();
    for (int st = 128; st > 0; st >>= 1) {
        if (threadIdx.x < st) {
            r0[threadIdx.x] += r0[threadIdx.x + st];
            r1[threadIdx.x] += r1[threadIdx.x + st];
        }
        __syncthreads();
    }
    if (threadIdx.x == 0) {
        ssrc[c * NN + n] = r0[0];
        sdst[c * NN + n] = r1[0];
    }
}

// monotonic float<->uint encoding for atomicMax over floats
__device__ __forceinline__ unsigned fenc(float f) {
    unsigned u = __float_as_uint(f);
    return (u & 0x80000000u) ? ~u : (u | 0x80000000u);
}
__device__ __forceinline__ float fdec(unsigned e) {
    unsigned u = (e & 0x80000000u) ? (e & 0x7FFFFFFFu) : ~e;
    return __uint_as_float(u);
}

// ---------------------------------------------------------------------------
// K4a: edge score (leaky relu) + segment max via atomicMax
// ---------------------------------------------------------------------------
__global__ __launch_bounds__(256)
void edge_score_max(const int* __restrict__ ei, int E, int Etot,
                    const float* __restrict__ ssrc, const float* __restrict__ sdst,
                    float* __restrict__ earr, unsigned* __restrict__ emax)
{
    const int j = blockIdx.x * blockDim.x + threadIdx.x;
    const int c = blockIdx.y;
    if (j >= Etot) return;
    int s, d;
    if (j < E) { s = ei[j]; d = ei[E + j]; } else { s = d = j - E; }
    float e = ssrc[c * NN + s] + sdst[c * NN + d];
    e = (e > 0.f) ? e : SLOPE * e;
    earr[(size_t)c * Etot + j] = e;
    atomicMax(&emax[c * NN + d], fenc(e));
}

// ---------------------------------------------------------------------------
// K4b: exp(e - max) + segment sum
// ---------------------------------------------------------------------------
__global__ __launch_bounds__(256)
void edge_exp_sum(const int* __restrict__ ei, int E, int Etot,
                  const float* __restrict__ earr, const unsigned* __restrict__ emax,
                  float* __restrict__ exarr, float* __restrict__ denom)
{
    const int j = blockIdx.x * blockDim.x + threadIdx.x;
    const int c = blockIdx.y;
    if (j >= Etot) return;
    int d = (j < E) ? ei[E + j] : (j - E);
    float ex = expf(earr[(size_t)c * Etot + j] - fdec(emax[c * NN + d]));
    exarr[(size_t)c * Etot + j] = ex;
    atomicAdd(&denom[c * NN + d], ex);
}

// ---------------------------------------------------------------------------
// K4c: scatter  g[dst][c] += alpha * h[c][src]   (g = "out" region, zeroed)
// grid (Etot, 2), block 256
// ---------------------------------------------------------------------------
__global__ __launch_bounds__(256)
void scatter_kernel(const int* __restrict__ ei, int E,
                    const float* __restrict__ h,
                    const float* __restrict__ exarr, int Etot,
                    const float* __restrict__ denom,
                    float* __restrict__ g)
{
    const int j = blockIdx.x, c = blockIdx.y;
    int s, d;
    if (j < E) { s = ei[j]; d = ei[E + j]; } else { s = d = j - E; }
    const float alpha = exarr[(size_t)c * Etot + j] / denom[c * NN + d];
    const float* hs = h + ((size_t)c * NN + s) * FF;
    float* gd = g + ((size_t)d * 2 + c) * FF;
    for (int f = threadIdx.x; f < FF; f += blockDim.x)
        atomicAdd(&gd[f], alpha * hs[f]);
}

// ---------------------------------------------------------------------------
// K5a: out1[n][c][l] = g[n][c][(l+s)%L] + bias[(l+s)%L]   (reverse roll)
// ---------------------------------------------------------------------------
__global__ __launch_bounds__(256)
void finalize_out1(const float* __restrict__ g,
                   const float* __restrict__ dtp, const float* __restrict__ dts,
                   const float* __restrict__ bias1, const float* __restrict__ bias2,
                   float* __restrict__ out1)
{
    const int n = blockIdx.x, c = blockIdx.y;
    const float* bias = (c == 0) ? bias1 : bias2;
    const int s = (int)(((c == 0) ? dtp[n] : dts[n]) * (float)FF);
    const float* gr = g + ((size_t)n * 2 + c) * FF;
    float* o = out1 + ((size_t)n * 2 + c) * FF;
    for (int l = threadIdx.x; l < FF; l += blockDim.x) {
        int idx = l + s; if (idx >= FF) idx -= FF;
        o[l] = gr[idx] + bias[idx];
    }
}

// ---------------------------------------------------------------------------
// K5b: out[n][c][l] += bias[l]  (in place, after out1 is done)
// ---------------------------------------------------------------------------
__global__ __launch_bounds__(256)
void add_bias_out(float* __restrict__ g,
                  const float* __restrict__ bias1, const float* __restrict__ bias2)
{
    const int n = blockIdx.x, c = blockIdx.y;
    const float* bias = (c == 0) ? bias1 : bias2;
    float* gr = g + ((size_t)n * 2 + c) * FF;
    for (int l = threadIdx.x; l < FF; l += blockDim.x)
        gr[l] += bias[l];
}

// ---------------------------------------------------------------------------
extern "C" void kernel_launch(void* const* d_in, const int* in_sizes, int n_in,
                              void* d_out, int out_size, void* d_ws, size_t ws_size,
                              hipStream_t stream)
{
    const float* x       = (const float*)d_in[0];
    const float* dtp     = (const float*)d_in[1];
    const float* dts     = (const float*)d_in[2];
    const float* conv1_w = (const float*)d_in[3];
    const float* conv1_b = (const float*)d_in[4];
    const float* bn1_g   = (const float*)d_in[5];
    const float* bn1_b   = (const float*)d_in[6];
    const float* bn1_m   = (const float*)d_in[7];
    const float* bn1_v   = (const float*)d_in[8];
    const float* conv2_w = (const float*)d_in[9];
    const float* conv2_b = (const float*)d_in[10];
    const float* bn2_g   = (const float*)d_in[11];
    const float* bn2_b   = (const float*)d_in[12];
    const float* bn2_m   = (const float*)d_in[13];
    const float* bn2_v   = (const float*)d_in[14];
    const float* gat1_w  = (const float*)d_in[15];
    const float* gat1_as = (const float*)d_in[16];
    const float* gat1_ad = (const float*)d_in[17];
    const float* gat1_bi = (const float*)d_in[18];
    const float* gat2_w  = (const float*)d_in[19];
    const float* gat2_as = (const float*)d_in[20];
    const float* gat2_ad = (const float*)d_in[21];
    const float* gat2_bi = (const float*)d_in[22];
    const int*   ei      = (const int*)d_in[23];

    const int E    = in_sizes[23] / 2;   // 32768
    const int Etot = E + NN;             // + self loops

    const size_t REG = (size_t)NN * 2 * FF;       // floats per output region
    float* out1  = (float*)d_out;                 // region 0
    float* outg  = out1 + REG;                    // region 1 (GAT out, scatter acc)
    float* tempv = outg + REG;                    // region 2 (temp_value)

    // workspace layout (bytes)
    char* ws = (char*)d_ws;
    const size_t OFF_TVBF  = 0;                                         // 2*N*F bf16
    const size_t OFF_WT    = OFF_TVBF + (size_t)2 * NN * FF * 2;        // 2*F*F bf16
    const size_t OFF_H     = OFF_WT   + (size_t)2 * FF * FF * 2;        // 2*N*F f32
    const size_t OFF_SSRC  = OFF_H    + (size_t)2 * NN * FF * 4;
    const size_t OFF_SDST  = OFF_SSRC + (size_t)2 * NN * 4;
    const size_t OFF_EMAX  = OFF_SDST + (size_t)2 * NN * 4;
    const size_t OFF_DENOM = OFF_EMAX + (size_t)2 * NN * 4;
    const size_t OFF_E     = OFF_DENOM + (size_t)2 * NN * 4;
    const size_t OFF_EX    = OFF_E    + (size_t)2 * Etot * 4;

    unsigned short* tvbf  = (unsigned short*)(ws + OFF_TVBF);
    unsigned short* wtbf  = (unsigned short*)(ws + OFF_WT);
    float*          h     = (float*)(ws + OFF_H);
    float*          ssrc  = (float*)(ws + OFF_SSRC);
    float*          sdst  = (float*)(ws + OFF_SDST);
    unsigned*       emax  = (unsigned*)(ws + OFF_EMAX);
    float*          denom = (float*)(ws + OFF_DENOM);
    float*          earr  = (float*)(ws + OFF_E);
    float*          exarr = (float*)(ws + OFF_EX);

    // zero accumulators (capture-safe)
    hipMemsetAsync(outg,  0, REG * sizeof(float), stream);
    hipMemsetAsync(emax,  0, (size_t)2 * NN * 4,  stream);
    hipMemsetAsync(denom, 0, (size_t)2 * NN * 4,  stream);

    // K0: weight convert+transpose
    wcvt_kernel<<<dim3(FF / 32, FF / 32, 2), dim3(32, 8), 0, stream>>>(gat1_w, gat2_w, wtbf);

    // K1: conv block + roll
    conv_roll_kernel<<<NN, 256, 0, stream>>>(x, dtp, dts,
        conv1_w, conv1_b, bn1_g, bn1_b, bn1_m, bn1_v,
        conv2_w, conv2_b, bn2_g, bn2_b, bn2_m, bn2_v,
        tempv, tvbf);

    // K2: WMMA GEMMs (both channels via grid.z)
    gemm_wmma_kernel<<<dim3(FF / 128, NN / 64, 2), 256, 0, stream>>>(tvbf, wtbf, h);

    // K3: attention projections
    dot_kernel<<<dim3(NN, 2), 256, 0, stream>>>(h, gat1_as, gat1_ad, gat2_as, gat2_ad, ssrc, sdst);

    // K4: edge softmax + scatter
    const int egrid = (Etot + 255) / 256;
    edge_score_max<<<dim3(egrid, 2), 256, 0, stream>>>(ei, E, Etot, ssrc, sdst, earr, emax);
    edge_exp_sum  <<<dim3(egrid, 2), 256, 0, stream>>>(ei, E, Etot, earr, emax, exarr, denom);
    scatter_kernel<<<dim3(Etot, 2), 256, 0, stream>>>(ei, E, h, exarr, Etot, denom, outg);

    // K5: outputs
    finalize_out1<<<dim3(NN, 2), 256, 0, stream>>>(outg, dtp, dts, gat1_bi, gat2_bi, out1);
    add_bias_out <<<dim3(NN, 2), 256, 0, stream>>>(outg, gat1_bi, gat2_bi);
}
